// FourierAttention_80556406604650
// MI455X (gfx1250) — matrix operand order repaired and checked by reading, not compiled, not run on hardware
//
#include <hip/hip_runtime.h>

// Shapes (fixed by reference): B=16, H=8, L=4096, E=64, M=64 -> BH=128, 2M=128
//
// Pipeline:
//   0) basis build: Fb[128,4096] f16 (rows cos / -sin), Bi[4096,128] f16 (transpose)
//   1) xT[bh][e][l] f16 = transpose+convert of x (LDS-tiled)
//   2) X[128,64]  = Fb @ x_bh          (WMMA, contiguous A and B fragments)
//   3) YT[o][c]   = complex mix * c_k/L (coalesced W float2 reads)
//   4) Out[4096,64] = Bi @ Y            (WMMA; Y staged in LDS via TDM tensor_load_to_lds)
//
// ws: Fb 1MB | Bi 1MB | X 4MB | YT 2MB | xT 64MB   (72 MB)

typedef __attribute__((ext_vector_type(16))) _Float16 v16h;
typedef __attribute__((ext_vector_type(8)))  _Float16 v8h;
typedef __attribute__((ext_vector_type(8)))  float    v8f;
typedef __attribute__((ext_vector_type(4)))  unsigned int v4u;
typedef __attribute__((ext_vector_type(8)))  int      v8i;
typedef __attribute__((ext_vector_type(4)))  int      v4i;

#define L_DIM 4096
#define E_DIM 64
#define M_DIM 64
#define BH_DIM 128
#define TWO_M 128

// ---------------------------------------------------------------------------
// Kernel 0: f16 DFT basis tables, exact integer phase reduction (L power of 2)
// ---------------------------------------------------------------------------
__global__ void fno_build_basis(const int* __restrict__ index,
                                _Float16* __restrict__ Fb,
                                _Float16* __restrict__ Bi) {
    const int c = blockIdx.x;            // 0..127
    const int k = index[c & (M_DIM - 1)];
    const bool isImag = (c >= M_DIM);
    const float w0 = 1.5339807878856412e-3f;  // 2*pi / 4096
    for (int l = threadIdx.x; l < L_DIM; l += blockDim.x) {
        const int p = (k * l) & (L_DIM - 1);
        float s, cv;
        __sincosf((float)p * w0, &s, &cv);
        const float val = isImag ? -s : cv;
        Fb[c * L_DIM + l] = (_Float16)val;
        Bi[l * TWO_M + c] = (_Float16)val;
    }
}

// ---------------------------------------------------------------------------
// Kernel 1: xT[bh][e][l] f16 = transpose(convert(x[bh][l][e])), 64x64 LDS tiles
// grid = 128 bh * 64 l-tiles, block = 256
// ---------------------------------------------------------------------------
__global__ __launch_bounds__(256)
void fno_cvt_xT(const float* __restrict__ x, _Float16* __restrict__ xT) {
    __shared__ _Float16 tile[64][72];   // padded rows (144B) for bank spread
    const int bh = blockIdx.x >> 6;
    const int l0 = (blockIdx.x & 63) * 64;
    const int t  = threadIdx.x;

    const int col  = t & 63;            // e
    const int rgrp = t >> 6;            // 0..3
#pragma unroll
    for (int i = 0; i < 16; ++i) {
        const int r = rgrp * 16 + i;    // l within tile
        tile[col][r] =
            (_Float16)x[((size_t)bh * L_DIM + l0 + r) * E_DIM + col];
    }
    __syncthreads();

    const int e = t >> 2;               // 0..63
    const int chunk = t & 3;            // 16 halves each
    v8h p0 = *(const v8h*)&tile[e][chunk * 16];
    v8h p1 = *(const v8h*)&tile[e][chunk * 16 + 8];
    v16h v = __builtin_shufflevector(p0, p1, 0, 1, 2, 3, 4, 5, 6, 7,
                                             8, 9, 10, 11, 12, 13, 14, 15);
    *(v16h*)(xT + ((size_t)bh * E_DIM + e) * L_DIM + l0 + chunk * 16) = v;
}

// ---------------------------------------------------------------------------
// Kernel 2: forward partial DFT.  X = Fb[128,4096] @ x_bh[4096,64].
// grid = 128 bh * 4, block = 8 waves; wave = one 16x16 tile, K loop 4096/32.
// All fragments are contiguous 16/32-byte loads.
// ---------------------------------------------------------------------------
__global__ __launch_bounds__(256)
void fno_fwd_dft(const _Float16* __restrict__ xT,
                 const _Float16* __restrict__ Fb,
                 float* __restrict__ X) {
    const int bh    = blockIdx.x >> 2;
    const int wave  = threadIdx.x >> 5;
    const int lane  = threadIdx.x & 31;
    const int tile  = ((blockIdx.x & 3) << 3) + wave;   // 0..31
    const int mtile = tile >> 2;                        // 0..7
    const int ntile = tile & 3;                         // 0..3
    const int col   = lane & 15;
    const int hi    = lane >> 4;

    const _Float16* __restrict__ arow = Fb + (mtile * 16 + col) * L_DIM;
    const _Float16* __restrict__ brow =
        xT + ((size_t)bh * E_DIM + ntile * 16 + col) * L_DIM;

    v8f acc = {};
    for (int k0 = 0; k0 < L_DIM; k0 += 32) {
        // A: lane<16 holds K 0..15, lane>=16 holds 16..31
        v16h A = *(const v16h*)(arow + k0 + hi * 16);
        // B: lane half selects K 0..15 / 16..31, ascending -> contiguous 32B
        v16h Bv = *(const v16h*)(brow + k0 + hi * 16);
        acc = __builtin_amdgcn_wmma_f32_16x16x32_f16(false, A, false, Bv,
                                                     (short)0, acc, false, false);
    }
    float* cp = X + (size_t)bh * (TWO_M * E_DIM)
                  + (mtile * 16) * E_DIM + ntile * 16 + col;
#pragma unroll
    for (int i = 0; i < 8; ++i) cp[(i + hi * 8) * E_DIM] = acc[i];
}

// ---------------------------------------------------------------------------
// Kernel 3: per-mode complex mix, writes Y TRANSPOSED: YT[bh][o][c], c=k|64+k.
// grid = 128 bh * 64 o, block = 64 threads (thread = mode k).
// W reads: lanes k consecutive -> contiguous float2. irfft scale c_k/L folded.
// ---------------------------------------------------------------------------
__global__ __launch_bounds__(64)
void fno_mix(const float* __restrict__ X,
             const float* __restrict__ W,     // [E][E][M][2]
             const int* __restrict__ index,
             _Float16* __restrict__ YT) {
    const int bh = blockIdx.x >> 6;
    const int o  = blockIdx.x & 63;
    const int k  = threadIdx.x;

    const float* xb = X + (size_t)bh * (TWO_M * E_DIM);
    const float2* W2 = (const float2*)W;

    float yr = 0.f, yi = 0.f;
#pragma unroll 4
    for (int e = 0; e < E_DIM; ++e) {
        const float xr = xb[k * E_DIM + e];
        const float xi = xb[(M_DIM + k) * E_DIM + e];
        const float2 wv = W2[(e * E_DIM + o) * M_DIM + k];
        yr = fmaf(xr, wv.x, fmaf(-xi, wv.y, yr));
        yi = fmaf(xr, wv.y, fmaf(xi, wv.x, yi));
    }
    const float s = (index[k] == 0 ? 1.0f : 2.0f) * (1.0f / (float)L_DIM);
    _Float16* yb = YT + (size_t)bh * (TWO_M * E_DIM);
    yb[o * TWO_M + k]         = (_Float16)(yr * s);
    yb[o * TWO_M + M_DIM + k] = (_Float16)(yi * s);
}

// ---------------------------------------------------------------------------
// Kernel 4: inverse partial DFT.  Out = Bi[4096,128] @ Y[128,64].
// Y_bh (16KB, transposed layout) staged into LDS by the Tensor Data Mover,
// B fragments then read as contiguous 32B ds loads.
// grid = 128 bh * 128, block = 8 waves; wave = one 16x16 tile, K loop 128/32.
// ---------------------------------------------------------------------------
__global__ __launch_bounds__(256)
void fno_inv_dft(const _Float16* __restrict__ Bi,
                 const _Float16* __restrict__ YT,
                 float* __restrict__ out) {
    __shared__ _Float16 sY[TWO_M * E_DIM];   // 16 KB, sole LDS object -> offset 0

    const int bh    = blockIdx.x >> 7;
    const int blk   = blockIdx.x & 127;
    const int wave  = threadIdx.x >> 5;
    const int lane  = threadIdx.x & 31;
    const int tile  = (blk << 3) + wave;     // 0..1023
    const int mtile = tile >> 2;             // 0..255 (l tiles)
    const int ntile = tile & 3;              // 0..3   (o tiles)
    const int col   = lane & 15;
    const int hi    = lane >> 4;

    if (wave == 0) {
        // TDM: copy YT_bh (2048 x 8B, contiguous) -> LDS offset 0.
        const unsigned long long ga =
            (unsigned long long)(YT + (size_t)bh * (TWO_M * E_DIM));
        v4u g0 = { 1u,                                   // count=1, user desc
                   0u,                                   // lds_addr = 0
                   (unsigned)(ga & 0xffffffffu),         // global_addr[31:0]
                   (unsigned)((ga >> 32) & 0x01ffffffu)  // global_addr[56:32]
                       | (2u << 30) };                   // type=2 (image)
        v8i g1 = { 0x00030000,   // data_size=3 (8B units), no mask/flags
                   0x08000000,   // tensor_dim0[15:0]=2048 in [31:16]
                   0x00010000,   // tensor_dim1=1 in [31:16]
                   0x08000000,   // tile_dim0=2048 in [31:16]
                   1,            // tile_dim1=1
                   2048,         // tensor_dim0_stride[31:0]
                   0, 0 };
        v4i gz4 = { 0, 0, 0, 0 };
        v8i gz8 = { 0, 0, 0, 0, 0, 0, 0, 0 };
        __builtin_amdgcn_tensor_load_to_lds(g0, g1, gz4, gz4, gz8, 0);
        __builtin_amdgcn_s_wait_tensorcnt(0);
    }
    __syncthreads();

    const _Float16* __restrict__ arow = Bi + (mtile * 16 + col) * TWO_M;
    const _Float16* brow = sY + (ntile * 16 + col) * TWO_M;

    v8f acc = {};
#pragma unroll
    for (int k0 = 0; k0 < TWO_M; k0 += 32) {
        v16h A = *(const v16h*)(arow + k0 + hi * 16);
        v16h Bv = *(const v16h*)(brow + k0 + hi * 16);   // LDS, 32B aligned
        acc = __builtin_amdgcn_wmma_f32_16x16x32_f16(false, A, false, Bv,
                                                     (short)0, acc, false, false);
    }
    float* op = out + (size_t)bh * L_DIM * E_DIM
                    + (mtile * 16) * E_DIM + ntile * 16 + col;
#pragma unroll
    for (int i = 0; i < 8; ++i) op[(i + hi * 8) * E_DIM] = acc[i];
}

// ---------------------------------------------------------------------------
extern "C" void kernel_launch(void* const* d_in, const int* in_sizes, int n_in,
                              void* d_out, int out_size, void* d_ws, size_t ws_size,
                              hipStream_t stream) {
    const float* x     = (const float*)d_in[0];   // [16,8,4096,64] f32
    const float* w     = (const float*)d_in[1];   // [64,64,64,2]   f32
    const int*   index = (const int*)d_in[2];     // [64] i32
    float*       out   = (float*)d_out;           // [16,8,4096,64] f32

    char* ws = (char*)d_ws;
    _Float16* Fb = (_Float16*)(ws);                 // 1 MB: [128][4096] f16
    _Float16* Bi = (_Float16*)(ws + (1u << 20));    // 1 MB: [4096][128] f16
    float*    X  = (float*)   (ws + (2u << 20));    // 4 MB: [128][128][64] f32
    _Float16* YT = (_Float16*)(ws + (6u << 20));    // 2 MB: [128][64][128] f16
    _Float16* xT = (_Float16*)(ws + (8u << 20));    // 64 MB: [128][64][4096] f16

    fno_build_basis<<<TWO_M, 256, 0, stream>>>(index, Fb, Bi);
    fno_cvt_xT<<<BH_DIM * 64, 256, 0, stream>>>(x, xT);
    fno_fwd_dft<<<BH_DIM * 4, 256, 0, stream>>>(xT, Fb, X);
    fno_mix<<<BH_DIM * M_DIM, 64, 0, stream>>>(X, w, index, YT);
    fno_inv_dft<<<BH_DIM * 128, 256, 0, stream>>>(Bi, YT, out);
}
